// AffinityConditionedAggregation_85152021611055
// MI455X (gfx1250) — matrix operand order repaired and checked by reading, not compiled, not run on hardware
//
#include <hip/hip_runtime.h>
#include <hip/hip_bf16.h>
#include <math.h>

typedef __attribute__((ext_vector_type(16))) _Float16 v16h;
typedef __attribute__((ext_vector_type(8)))  float    v8f;

#define NF 64
#define RTILE 1024

// ---------- order-preserving float <-> uint (for atomicMax pooling) ----------
__device__ __forceinline__ unsigned enc_f32(float f) {
  unsigned u = __float_as_uint(f);
  return (u & 0x80000000u) ? ~u : (u | 0x80000000u);
}
__device__ __forceinline__ float dec_f32(unsigned e) {
  return (e & 0x80000000u) ? __uint_as_float(e & 0x7FFFFFFFu)
                           : __uint_as_float(~e);
}

// ---------- edge list with self loops ----------
__global__ void k_build_edges(const int* __restrict__ ei, int* __restrict__ row,
                              int* __restrict__ col, long long E, long long ET) {
  long long e = (long long)blockIdx.x * blockDim.x + threadIdx.x;
  if (e >= ET) return;
  if (e < E) { row[e] = ei[e]; col[e] = ei[E + e]; }
  else       { int i = (int)(e - E); row[e] = i; col[e] = i; }
}

// ---------- helpers for the WMMA fragment build (all-constant vector indices) ----
__device__ __forceinline__ float4 diff4(const float* __restrict__ a,
                                        const float* __restrict__ b, int o) {
  float4 u = *(const float4*)(a + o);
  float4 v = *(const float4*)(b + o);
  return make_float4(u.x - v.x, u.y - v.y, u.z - v.z, u.w - v.w);
}
__device__ __forceinline__ void pack8(v16h& dst, const int base, float4 p, float4 q) {
  dst[base + 0] = (_Float16)p.x; dst[base + 1] = (_Float16)p.y;
  dst[base + 2] = (_Float16)p.z; dst[base + 3] = (_Float16)p.w;
  dst[base + 4] = (_Float16)q.x; dst[base + 5] = (_Float16)q.y;
  dst[base + 6] = (_Float16)q.z; dst[base + 7] = (_Float16)q.w;
}

// ---------- per-edge sqdist via WMMA Gram diagonal; aff = exp(-v2*sqdist) ----------
// One wave32 handles 16 edges. A = diff matrix (16x32 f16), B = diff^T (32x16),
// diag(A*B) = ||x_r - x_c||^2 per edge, f32 accumulate over two K=32 chunks.
// Lane-dependent slice selection is folded into the load ADDRESSES so every
// vector-register subscript is a compile-time constant (no spills, no cndmask).
__global__ void k_aff_wmma(const float* __restrict__ x, const int* __restrict__ row,
                           const int* __restrict__ col, const float* __restrict__ v2p,
                           float* __restrict__ aff, float* __restrict__ losssum,
                           long long ET) {
  const int lane = threadIdx.x & 31;
  const long long wave = ((long long)blockIdx.x * blockDim.x + threadIdx.x) >> 5;
  const long long e0 = wave * 16;
  if (e0 >= ET) return;                       // wave-uniform: EXEC stays all-ones
  const int m = lane & 15;
  long long e = e0 + m;
  if (e >= ET) e = ET - 1;                    // clamp (ET need not be a multiple of 16)
  const float* xr = x + (long long)row[e] * NF;
  const float* xc = x + (long long)col[e] * NF;
  const int alo = (lane < 16) ? 0 : 16;  // A: feats alo..alo+15
  const int blo = (lane < 16) ? 0 : 16;  // B: feats blo..blo+15

  v8f acc = {};
#pragma unroll
  for (int q = 0; q < 2; ++q) {
    const int f0 = q * 32;
    // A fragment (16-bit A 16x32 layout)
    float4 a0 = diff4(xr, xc, f0 + alo);
    float4 a1 = diff4(xr, xc, f0 + alo + 4);
    float4 a2 = diff4(xr, xc, f0 + alo + 8);
    float4 a3 = diff4(xr, xc, f0 + alo + 12);
    v16h A;
    pack8(A, 0, a0, a1);
    pack8(A, 8, a2, a3);
    // B fragment (16-bit B 32x16 layout): contiguous K run per half-wave
    float4 b0 = diff4(xr, xc, f0 + blo);
    float4 b1 = diff4(xr, xc, f0 + blo + 4);
    float4 b2 = diff4(xr, xc, f0 + blo + 8);
    float4 b3 = diff4(xr, xc, f0 + blo + 12);
    v16h B;
    pack8(B, 0, b0, b1);
    pack8(B, 8, b2, b3);
    acc = __builtin_amdgcn_wmma_f32_16x16x32_f16(false, A, false, B, (short)0, acc,
                                                 false, false);
  }
  // diagonal: m<8 -> (lane m, elem m); m>=8 -> (lane m+16, elem m-8)
  bool act = false; long long eo = 0; int ci = 0;
  if (lane < 8)        { act = true; eo = e0 + lane;        ci = lane; }
  else if (lane >= 24) { act = true; eo = e0 + (lane - 16); ci = lane - 24; }
  float sq = 0.f;
#pragma unroll
  for (int t = 0; t < 8; ++t) sq = (ci == t) ? acc[t] : sq;
  if (sq < 0.f) sq = 0.f;
  float part = (act && eo < ET) ? sq : 0.f;
#pragma unroll
  for (int o = 16; o > 0; o >>= 1) part += __shfl_down(part, o, 32);
  if (lane == 0) atomicAdd(losssum, part);
  if (act && eo < ET) aff[eo] = __expf(-v2p[0] * sq);
}

__global__ void k_loss(const float* __restrict__ losssum, float* __restrict__ out,
                       long long ET) {
  out[0] = losssum[0] / (float)ET;
}

// ---------- CSR build ----------
__global__ void k_deg(const int* __restrict__ row, int* __restrict__ deg, long long ET) {
  long long e = (long long)blockIdx.x * blockDim.x + threadIdx.x;
  if (e < ET) atomicAdd(&deg[row[e]], 1);
}

__global__ void k_scan1(const int* __restrict__ in, int* __restrict__ out,
                        int* __restrict__ bsums, int n) {
  __shared__ int s[256];
  int g = blockIdx.x * 256 + threadIdx.x;
  int v = (g < n) ? in[g] : 0;
  s[threadIdx.x] = v; __syncthreads();
  for (int o = 1; o < 256; o <<= 1) {
    int t = (threadIdx.x >= o) ? s[threadIdx.x - o] : 0;
    __syncthreads(); s[threadIdx.x] += t; __syncthreads();
  }
  if (g < n) out[g] = s[threadIdx.x];
  if (threadIdx.x == 255) bsums[blockIdx.x] = s[255];
}
__global__ void k_scan2(int* __restrict__ bsums, int nb) {
  __shared__ int s[256];
  int v = (threadIdx.x < nb) ? bsums[threadIdx.x] : 0;
  s[threadIdx.x] = v; __syncthreads();
  for (int o = 1; o < 256; o <<= 1) {
    int t = (threadIdx.x >= o) ? s[threadIdx.x - o] : 0;
    __syncthreads(); s[threadIdx.x] += t; __syncthreads();
  }
  if (threadIdx.x < nb) bsums[threadIdx.x] = s[threadIdx.x];
}
__global__ void k_scan3(int* __restrict__ out, const int* __restrict__ bsums, int n) {
  int g = blockIdx.x * 256 + threadIdx.x;
  if (g < n && blockIdx.x > 0) out[g] += bsums[blockIdx.x - 1];
}

__global__ void k_rowptr(const int* __restrict__ S, int* __restrict__ rowptr,
                         int* __restrict__ cursor, int N) {
  int i = blockIdx.x * blockDim.x + threadIdx.x;
  if (i >= N) return;
  int prev = (i > 0) ? S[i - 1] : 0;
  rowptr[i] = prev; cursor[i] = prev;
  if (i == N - 1) rowptr[N] = S[N - 1];
}

__global__ void k_csrfill(const int* __restrict__ row, const int* __restrict__ col,
                          const float* __restrict__ aff, int* __restrict__ cursor,
                          int* __restrict__ csr_col, float* __restrict__ csr_aff,
                          long long ET) {
  long long e = (long long)blockIdx.x * blockDim.x + threadIdx.x;
  if (e >= ET) return;
  int p = atomicAdd(&cursor[row[e]], 1);
  csr_col[p] = col[e]; csr_aff[p] = aff[e];
}

// ---------- label propagation (Jacobi, double buffered) ----------
__global__ void k_iota(int* __restrict__ lab, int N) {
  int i = blockIdx.x * blockDim.x + threadIdx.x;
  if (i < N) lab[i] = i;
}
__global__ void k_lp(const int* __restrict__ rowptr, const int* __restrict__ csr_col,
                     const float* __restrict__ csr_aff, const int* __restrict__ lin,
                     int* __restrict__ lout, int N) {
  int i = blockIdx.x * blockDim.x + threadIdx.x;
  if (i >= N) return;
  int a = rowptr[i], b = rowptr[i + 1];
  float best_s = -__builtin_inff(); int best_l = N;
  for (int ii = a; ii < b; ++ii) {
    int li = lin[csr_col[ii]];
    float s = 0.f;
    for (int jj = a; jj < b; ++jj) {
      int lj = lin[csr_col[jj]];
      s += (lj == li) ? csr_aff[jj] : 0.f;
    }
    if (s > best_s || (s == best_s && li < best_l)) { best_s = s; best_l = li; }
  }
  lout[i] = best_l;
}

// ---------- dense relabel ----------
__global__ void k_present(const int* __restrict__ lab, int* __restrict__ present, int N) {
  int i = blockIdx.x * blockDim.x + threadIdx.x;
  if (i < N) present[lab[i]] = 1;
}
__global__ void k_relabel(const int* __restrict__ S, const int* __restrict__ lab,
                          int* __restrict__ clab, float* __restrict__ out_clab, int N) {
  int i = blockIdx.x * blockDim.x + threadIdx.x;
  if (i >= N) return;
  int c = S[lab[i]] - 1;
  clab[i] = c; out_clab[i] = (float)c;
}

// ---------- max pool ----------
__global__ void k_pool(const float* __restrict__ x, const int* __restrict__ clab,
                       const int* __restrict__ batch, unsigned* __restrict__ cxu,
                       int* __restrict__ cnt, int* __restrict__ cbi, int N) {
  long long idx = (long long)blockIdx.x * blockDim.x + threadIdx.x;
  if (idx >= (long long)N * NF) return;
  int i = (int)(idx >> 6), f = (int)(idx & 63);
  int c = clab[i];
  atomicMax(&cxu[(long long)c * NF + f], enc_f32(x[idx]));
  if (f == 0) { atomicAdd(&cnt[c], 1); atomicMax(&cbi[c], batch[i]); }
}
__global__ void k_fin_cx(const unsigned* __restrict__ cxu, const int* __restrict__ cnt,
                         float* __restrict__ out_cx, int N) {
  long long idx = (long long)blockIdx.x * blockDim.x + threadIdx.x;
  if (idx >= (long long)N * NF) return;
  int r = (int)(idx >> 6);
  out_cx[idx] = (cnt[r] > 0) ? dec_f32(cxu[idx]) : 0.f;
}
__global__ void k_cb(const int* __restrict__ cbi, float* __restrict__ out_cb, int N) {
  int i = blockIdx.x * blockDim.x + threadIdx.x;
  if (i < N) out_cb[i] = (float)cbi[i];
}

// ---------- coalesce: pack keys, 4x8-bit LSD counting sort, dedup ----------
__global__ void k_keys(const int* __restrict__ row, const int* __restrict__ col,
                       const int* __restrict__ clab, unsigned* __restrict__ keys,
                       long long ET) {
  long long e = (long long)blockIdx.x * blockDim.x + threadIdx.x;
  if (e >= ET) return;
  keys[e] = ((unsigned)clab[row[e]] << 16) | (unsigned)clab[col[e]];
}
__global__ void k_hist(const unsigned* __restrict__ keys, int* __restrict__ C,
                       long long ET, int shift, int RB) {
  __shared__ int h[256];
  h[threadIdx.x] = 0; __syncthreads();
  long long base = (long long)blockIdx.x * RTILE;
  for (int t = 0; t < 4; ++t) {
    long long g = base + threadIdx.x + t * 256;
    if (g < ET) atomicAdd(&h[(keys[g] >> shift) & 255], 1);
  }
  __syncthreads();
  C[threadIdx.x * RB + blockIdx.x] = h[threadIdx.x];
}
__global__ void k_offsets(int* __restrict__ C, int RB) {
  __shared__ int s[256];
  int t = threadIdx.x;
  int tot = 0;
  for (int b = 0; b < RB; ++b) tot += C[t * RB + b];
  s[t] = tot; __syncthreads();
  for (int o = 1; o < 256; o <<= 1) {
    int v = (t >= o) ? s[t - o] : 0;
    __syncthreads(); s[t] += v; __syncthreads();
  }
  int run = s[t] - tot;  // exclusive base for this bin
  for (int b = 0; b < RB; ++b) { int v = C[t * RB + b]; C[t * RB + b] = run; run += v; }
}
__global__ void k_scatter(const unsigned* __restrict__ in, unsigned* __restrict__ out,
                          int* __restrict__ C, long long ET, int shift, int RB) {
  long long base = (long long)blockIdx.x * RTILE;
  for (int t = 0; t < 4; ++t) {
    long long g = base + threadIdx.x + t * 256;
    if (g < ET) {
      unsigned k = in[g];
      int bin = (int)((k >> shift) & 255);
      int p = atomicAdd(&C[bin * RB + blockIdx.x], 1);
      out[p] = k;
    }
  }
}
__global__ void k_dedup(const unsigned* __restrict__ keys, float* __restrict__ out_cei,
                        long long ET) {
  long long e = (long long)blockIdx.x * blockDim.x + threadIdx.x;
  if (e >= ET) return;
  unsigned k = keys[e];
  bool dup = (e > 0) && (keys[e - 1] == k);
  out_cei[e]      = dup ? -1.f : (float)(k >> 16);
  out_cei[ET + e] = dup ? -1.f : (float)(k & 0xFFFFu);
}

extern "C" void kernel_launch(void* const* d_in, const int* in_sizes, int n_in,
                              void* d_out, int out_size, void* d_ws, size_t ws_size,
                              hipStream_t stream) {
  (void)n_in; (void)out_size; (void)ws_size;
  const float* x     = (const float*)d_in[0];
  const int*   ei    = (const int*)d_in[1];
  const int*   batch = (const int*)d_in[2];
  const float* v2    = (const float*)d_in[3];
  const int       N  = in_sizes[2];
  const long long E  = (long long)in_sizes[1] / 2;
  const long long ET = E + N;

  char* w = (char*)d_ws; size_t off = 0;
  auto walloc = [&](size_t bytes) -> void* {
    void* p = w + off; off = (off + bytes + 255) & ~(size_t)255; return p;
  };
  int*      row_e   = (int*)walloc((size_t)ET * 4);
  int*      col_e   = (int*)walloc((size_t)ET * 4);
  float*    aff     = (float*)walloc((size_t)ET * 4);
  int*      csr_col = (int*)walloc((size_t)ET * 4);
  float*    csr_aff = (float*)walloc((size_t)ET * 4);
  int*      deg     = (int*)walloc((size_t)N * 4);
  int*      S       = (int*)walloc((size_t)N * 4);
  int*      bsums   = (int*)walloc(1024 * 4);
  int*      rowptr  = (int*)walloc(((size_t)N + 1) * 4);
  int*      cursor  = (int*)walloc((size_t)N * 4);
  int*      labA    = (int*)walloc((size_t)N * 4);
  int*      labB    = (int*)walloc((size_t)N * 4);
  int*      present = (int*)walloc((size_t)N * 4);
  int*      clab    = (int*)walloc((size_t)N * 4);
  unsigned* cxu     = (unsigned*)walloc((size_t)N * NF * 4);
  int*      cnt     = (int*)walloc((size_t)N * 4);
  int*      cbi     = (int*)walloc((size_t)N * 4);
  float*    losss   = (float*)walloc(256);
  // buffers dead after LP are reused by the sort:
  unsigned* keysA = (unsigned*)csr_col;
  unsigned* keysB = (unsigned*)row_e;   // row_e dead after k_keys
  int*      radC  = (int*)csr_aff;      // RB*256*4 <= ET*4
  const int RB = (int)((ET + RTILE - 1) / RTILE);

  float* out_cx   = (float*)d_out;
  float* out_cei  = out_cx + (size_t)N * NF;
  float* out_cb   = out_cei + 2 * (size_t)ET;
  float* out_clab = out_cb + N;
  float* out_loss = out_clab + N;

  dim3 B256(256);
  auto g1 = [&](long long n) { return dim3((unsigned)((n + 255) / 256)); };
  const int SB = (N + 255) / 256;  // blocks for 256-wide scans

  hipMemsetAsync(deg, 0, (size_t)N * 4, stream);
  hipMemsetAsync(losss, 0, 4, stream);

  k_build_edges<<<g1(ET), B256, 0, stream>>>(ei, row_e, col_e, E, ET);
  // one wave32 per 16 edges -> 2 threads per edge
  k_aff_wmma<<<g1(ET * 2), B256, 0, stream>>>(x, row_e, col_e, v2, aff, losss, ET);
  k_loss<<<1, 1, 0, stream>>>(losss, out_loss, ET);

  k_deg<<<g1(ET), B256, 0, stream>>>(row_e, deg, ET);
  k_scan1<<<SB, B256, 0, stream>>>(deg, S, bsums, N);
  k_scan2<<<1, B256, 0, stream>>>(bsums, SB);
  k_scan3<<<SB, B256, 0, stream>>>(S, bsums, N);
  k_rowptr<<<g1(N), B256, 0, stream>>>(S, rowptr, cursor, N);
  k_csrfill<<<g1(ET), B256, 0, stream>>>(row_e, col_e, aff, cursor, csr_col, csr_aff, ET);

  k_iota<<<g1(N), B256, 0, stream>>>(labA, N);
  int* cur = labA; int* nxt = labB;
  for (int it = 0; it < 30; ++it) {
    k_lp<<<g1(N), B256, 0, stream>>>(rowptr, csr_col, csr_aff, cur, nxt, N);
    int* t = cur; cur = nxt; nxt = t;
  }

  hipMemsetAsync(present, 0, (size_t)N * 4, stream);
  k_present<<<g1(N), B256, 0, stream>>>(cur, present, N);
  k_scan1<<<SB, B256, 0, stream>>>(present, S, bsums, N);
  k_scan2<<<1, B256, 0, stream>>>(bsums, SB);
  k_scan3<<<SB, B256, 0, stream>>>(S, bsums, N);
  k_relabel<<<g1(N), B256, 0, stream>>>(S, cur, clab, out_clab, N);

  hipMemsetAsync(cxu, 0, (size_t)N * NF * 4, stream);
  hipMemsetAsync(cnt, 0, (size_t)N * 4, stream);
  hipMemsetAsync(cbi, 0xFF, (size_t)N * 4, stream);   // -1
  k_pool<<<g1((long long)N * NF), B256, 0, stream>>>(x, clab, batch, cxu, cnt, cbi, N);
  k_fin_cx<<<g1((long long)N * NF), B256, 0, stream>>>(cxu, cnt, out_cx, N);
  k_cb<<<g1(N), B256, 0, stream>>>(cbi, out_cb, N);

  k_keys<<<g1(ET), B256, 0, stream>>>(row_e, col_e, clab, keysA, ET);
  const unsigned* rin = keysA; unsigned* rout = keysB;
  for (int p = 0; p < 4; ++p) {
    int shift = p * 8;
    k_hist<<<RB, B256, 0, stream>>>(rin, radC, ET, shift, RB);
    k_offsets<<<1, B256, 0, stream>>>(radC, RB);
    k_scatter<<<RB, B256, 0, stream>>>(rin, rout, radC, ET, shift, RB);
    unsigned* t = (unsigned*)rin; rin = rout; rout = t;
  }
  k_dedup<<<g1(ET), B256, 0, stream>>>(rin, out_cei, ET);
}
